// PersonaAwareGeneration_2259152798400
// MI455X (gfx1250) — matrix operand order repaired and checked by reading, not compiled, not run on hardware
//
#include <hip/hip_runtime.h>
#include <cstdint>
#include <cstddef>
#include <cmath>

// ---------------- model dimensions (compile-time) ----------------
#define D_   768
#define H_   8
#define B_   32
#define TC_  48
#define P_   16
#define NN_  64
#define TD_  32
#define V_   50257

// ---------------- WMMA GEMM tiling ----------------
#define BM   128    // M tile (8 waves x 16 rows)
#define BN   64     // N tile (4 accumulators of 16 per wave)
#define BK   32     // K step (one v_wmma_f32_16x16x32_bf16 per fragment pair)
#define LDST 40     // padded LDS row stride in bf16 elems (80B -> 16B aligned, conflict-free)

#define AS1 __attribute__((address_space(1)))
#define AS3 __attribute__((address_space(3)))

typedef __attribute__((ext_vector_type(16))) __bf16 v16bf;
typedef __attribute__((ext_vector_type(8)))  float  v8f;

// f32 -> bf16 round-to-nearest-even (bit level; no __bf16 arithmetic needed)
__device__ __forceinline__ unsigned short f2bf(float f) {
  unsigned int u = __float_as_uint(f);
  return (unsigned short)((u + 0x7FFFu + ((u >> 16) & 1u)) >> 16);
}
__device__ __forceinline__ unsigned int pack2bf(float a, float b) {
  return (unsigned int)f2bf(a) | ((unsigned int)f2bf(b) << 16);
}

// =====================================================================
// Generic batched bf16 WMMA GEMM:  C = act(alpha * A@B + bias)
//   A: [M,K] f32 row-major (lda)
//   B: flags&1 ? operand given as [N,K] row-major (B^T)  : [K,N] row-major
//   C: [M,N] f32 row-major (ldc)
//   batch z decomposed as z0 = z % Z0, z1 = z / Z0; per-axis strides.
//   flags: bit0 = B-transposed, bit1 = ReLU
// 256 threads = 8 wave32 waves. 128x64 C tile: wave w owns rows w*16..+15 and
// all 64 cols (4 x 16x16 accumulators). One A fragment feeds 4 WMMAs.
// Bounds handled by clamped addressing + select (no EXEC-predicated loads);
// uniform fast paths use aligned float4 (global_load_b128).
// =====================================================================
__global__ void __launch_bounds__(256)
gemm_bf16_wmma(const float* __restrict__ Abase, const float* __restrict__ Bbase,
               float* __restrict__ Cbase, const float* __restrict__ bias,
               int M, int N, int K, int lda, int ldb, int ldc, int Z0,
               long long sA0, long long sA1, long long sB0, long long sB1,
               long long sC0, long long sC1, float alpha, int flags)
{
  __shared__ unsigned short At[BM][LDST];   // A tile, [m][k] bf16
  __shared__ unsigned short Bt[BN][LDST];   // B tile, stored transposed [n][k] bf16

  const int z  = blockIdx.z;
  const int z0 = z % Z0;
  const int z1 = z / Z0;
  const float* A  = Abase + (long long)z0 * sA0 + (long long)z1 * sA1;
  const float* Bm = Bbase + (long long)z0 * sB0 + (long long)z1 * sB1;
  float*       C  = Cbase + (long long)z0 * sC0 + (long long)z1 * sC1;

  const int tid  = threadIdx.x;
  const int lane = tid & 31;
  const int wave = tid >> 5;          // 0..7 -> row group
  const int m0   = blockIdx.y * BM;
  const int n0   = blockIdx.x * BN;
  const bool bt  = (flags & 1) != 0;

  v8f acc[4];
#pragma unroll
  for (int g = 0; g < 4; ++g)
#pragma unroll
    for (int i = 0; i < 8; ++i) acc[g][i] = 0.0f;

  // staging index precompute
  const int arA = tid >> 1;           // 0..127 (A row)
  const int acA = (tid & 1) * 16;     // 0/16   (A k start, 16 elems)
  const int arB = tid >> 2;           // 0..63  (B^T row = n)
  const int acB = (tid & 3) * 8;      // 0,8,16,24 (k start, 8 elems)
  const int kkB = tid >> 4;           // 0..15  (k row, [K,N] transpose path)
  const int nnB = (tid & 15) * 4;     // 0..60  (n start, 4 elems)

  const bool ldbAligned = ((ldb & 3) == 0);
  const int  kTiles = (K + BK - 1) / BK;

  for (int kt = 0; kt < kTiles; ++kt) {
    const int k0 = kt * BK;
    const bool kfull = (k0 + BK <= K);   // uniform

    // ---- stage A tile: At[m][k] = bf16(A[m0+m][k0+k]) ----
    {
      const int gmr = m0 + arA;
      const int gm  = (gmr < M) ? gmr : (M - 1);       // clamp, zero later
      const float rmask = (gmr < M) ? 1.0f : 0.0f;
      const float* Ap = A + (long long)gm * lda + k0 + acA;
      float v[16];
      if (kfull) {
        const float4* Ap4 = reinterpret_cast<const float4*>(Ap);
#pragma unroll
        for (int q = 0; q < 4; ++q) {
          const float4 t = Ap4[q];
          v[q*4+0] = t.x; v[q*4+1] = t.y; v[q*4+2] = t.z; v[q*4+3] = t.w;
        }
      } else {
#pragma unroll
        for (int j = 0; j < 16; ++j) {
          const bool ok = (k0 + acA + j) < K;
          const float t = Ap[ok ? j : 0];              // clamped addr, uncond load
          v[j] = ok ? t : 0.0f;
        }
      }
#pragma unroll
      for (int j = 0; j < 16; ++j) v[j] *= rmask;
      const uint4 p0 = make_uint4(pack2bf(v[0],v[1]),  pack2bf(v[2],v[3]),
                                  pack2bf(v[4],v[5]),  pack2bf(v[6],v[7]));
      const uint4 p1 = make_uint4(pack2bf(v[8],v[9]),  pack2bf(v[10],v[11]),
                                  pack2bf(v[12],v[13]),pack2bf(v[14],v[15]));
      *reinterpret_cast<uint4*>(&At[arA][acA])     = p0;
      *reinterpret_cast<uint4*>(&At[arA][acA + 8]) = p1;
    }
    // ---- stage B tile as Bt[n][k] ----
    if (bt) {
      // operand given as [N,K] row-major: direct copy
      const int gnr = n0 + arB;
      const int gn  = (gnr < N) ? gnr : (N - 1);
      const float nmask = (gnr < N) ? 1.0f : 0.0f;
      const float* Bp = Bm + (long long)gn * ldb + k0 + acB;
      float v[8];
      if (kfull && ldbAligned) {
        const float4* Bp4 = reinterpret_cast<const float4*>(Bp);
        const float4 t0 = Bp4[0], t1 = Bp4[1];
        v[0]=t0.x; v[1]=t0.y; v[2]=t0.z; v[3]=t0.w;
        v[4]=t1.x; v[5]=t1.y; v[6]=t1.z; v[7]=t1.w;
      } else {
#pragma unroll
        for (int j = 0; j < 8; ++j) {
          const bool ok = (k0 + acB + j) < K;
          const float t = Bp[ok ? j : 0];
          v[j] = ok ? t : 0.0f;
        }
      }
#pragma unroll
      for (int j = 0; j < 8; ++j) v[j] *= nmask;
      const uint4 pk = make_uint4(pack2bf(v[0],v[1]), pack2bf(v[2],v[3]),
                                  pack2bf(v[4],v[5]), pack2bf(v[6],v[7]));
      *reinterpret_cast<uint4*>(&Bt[arB][acB]) = pk;
    } else {
      // operand is [K,N] row-major: coalesced read along N, transpose into LDS
      const bool nfull = (n0 + BN <= N);   // uniform
#pragma unroll
      for (int p = 0; p < 2; ++p) {
        const int gkr = k0 + p * 16 + kkB;
        const int gk  = (gkr < K) ? gkr : (K - 1);
        const float kmask = (gkr < K) ? 1.0f : 0.0f;
        const float* Bp = Bm + (long long)gk * ldb + n0 + nnB;
        float v[4];
        if (nfull && ldbAligned) {
          const float4 t = *reinterpret_cast<const float4*>(Bp);
          v[0]=t.x; v[1]=t.y; v[2]=t.z; v[3]=t.w;
        } else {
#pragma unroll
          for (int j = 0; j < 4; ++j) {
            const bool ok = (n0 + nnB + j) < N;
            const float t = Bp[ok ? j : 0];
            v[j] = ok ? t : 0.0f;
          }
        }
#pragma unroll
        for (int j = 0; j < 4; ++j)
          Bt[nnB + j][p * 16 + kkB] = f2bf(v[j] * kmask);
      }
    }
    // ---- prefetch next K tile (global_prefetch_b8) ----
    if (k0 + BK < K) {
      const int gmp = m0 + arA; 
      __builtin_prefetch(&A[(long long)((gmp < M) ? gmp : (M-1)) * lda + k0 + BK + acA], 0, 1);
      if (bt) {
        const int gnp = n0 + arB;
        __builtin_prefetch(&Bm[(long long)((gnp < N) ? gnp : (N-1)) * ldb + k0 + BK + acB], 0, 1);
      } else {
        const int gkp = k0 + BK + kkB;
        __builtin_prefetch(&Bm[(long long)((gkp < K) ? gkp : (K-1)) * ldb + n0 + nnB], 0, 1);
      }
    }
    __syncthreads();

    // ---- fragments per ISA 7.12.2 layouts (two ds_load_b128 each) ----
    const int hf  = lane >> 4;   // 0/1 lane-half
    const int l16 = lane & 15;
    v16bf afrag;
    {
      // A (16x32): row m = wave*16 + l16;
      // lanes 0-15: K {0..7, 16..23}; lanes 16-31: K {8..15, 24..31}
      const unsigned short* ap = &At[wave * 16 + l16][hf * 8];
      reinterpret_cast<uint4*>(&afrag)[0] = *reinterpret_cast<const uint4*>(ap);
      reinterpret_cast<uint4*>(&afrag)[1] = *reinterpret_cast<const uint4*>(ap + 16);
    }
    v16bf bfrag[4];
#pragma unroll
    for (int g = 0; g < 4; ++g) {
      // B (32x16): col n = g*16 + l16; lanes 0-15: K {0..7, 16..23}; lanes 16-31: K {8..15, 24..31}
      const unsigned short* bp = &Bt[g * 16 + l16][hf * 8];
      reinterpret_cast<uint4*>(&bfrag[g])[0] = *reinterpret_cast<const uint4*>(bp);
      reinterpret_cast<uint4*>(&bfrag[g])[1] = *reinterpret_cast<const uint4*>(bp + 16);
    }
#pragma unroll
    for (int g = 0; g < 4; ++g)
      acc[g] = __builtin_amdgcn_wmma_f32_16x16x32_bf16(false, afrag, false, bfrag[g],
                                                       (short)0, acc[g], false, false);
    __syncthreads();
  }

  // ---- epilogue: VGPR e holds row (8*laneHalf + e), col = lane&15 ----
  const int hf  = lane >> 4;
  const int l16 = lane & 15;
  const int gmb = m0 + wave * 16 + hf * 8;
  const bool relu = (flags & 2) != 0;
#pragma unroll
  for (int g = 0; g < 4; ++g) {
    const int c = n0 + g * 16 + l16;
    if (c >= N) continue;
    const float bv = bias ? bias[c] : 0.0f;
#pragma unroll
    for (int e = 0; e < 8; ++e) {
      const int gm = gmb + e;
      if (gm < M) {
        float v = acc[g][e] * alpha + bv;
        if (relu) v = fmaxf(v, 0.0f);
        C[(long long)gm * ldc + c] = v;
      }
    }
  }
}

// =====================================================================
// Elementwise / reduction helper kernels
// =====================================================================

// out[b,t,:] = x[b,t,:] + pe_row[:]   (faithful quirk: single pe row added)
__global__ void add_pe_row(const float* __restrict__ x, const float* __restrict__ pe_row,
                           float* __restrict__ out, int total, int D)
{
  int i = blockIdx.x * blockDim.x + threadIdx.x;
  if (i >= total) return;
  out[i] = x[i] + pe_row[i % D];
}

// out[b,t,:] = emb[ids[b,t],:] + pe_row[:]
__global__ void embed_pe(const int* __restrict__ ids, const float* __restrict__ emb,
                         const float* __restrict__ pe_row, float* __restrict__ out,
                         int total, int D)
{
  int i = blockIdx.x * blockDim.x + threadIdx.x;
  if (i >= total) return;
  int r = i / D, d = i - r * D;
  out[i] = emb[(long long)ids[r] * D + d] + pe_row[d];
}

// G[b, 0:P] = persona[b];  G[b, P+t] = att[b,t] * !mask[b,t]
__global__ void concat_graph(const float* __restrict__ persona, const float* __restrict__ att,
                             const unsigned char* __restrict__ mask, float* __restrict__ G,
                             int total)
{
  int i = blockIdx.x * blockDim.x + threadIdx.x;
  if (i >= total) return;
  int d   = i % D_;
  int row = i / D_;
  int n   = row % NN_;
  int b   = row / NN_;
  float v;
  if (n < P_) {
    v = persona[((long long)b * P_ + n) * D_ + d];
  } else {
    int t = n - P_;
    v = att[((long long)b * TC_ + t) * D_ + d];
    if (mask[b * TC_ + t]) v = 0.0f;
  }
  G[i] = v;
}

// row-wise softmax with optional causal + key mask.
// scores layout: [(h*B + b)*Tq + t][Sdim]; one 64-thread block per row (Sdim<=64).
// Row is staged global->LDS with the CDNA5 async-copy path when available.
// Probe result: builtin exists with signature (global int*, local int*, imm, imm).
#if __has_builtin(__builtin_amdgcn_global_load_async_to_lds_b32) && \
    __has_builtin(__builtin_amdgcn_s_wait_asynccnt)
#define SOFTMAX_ASYNC 1
#else
#define SOFTMAX_ASYNC 0
#endif

__global__ void softmax_rows(float* __restrict__ S, const unsigned char* __restrict__ mask,
                             int Bdim, int Tq, int Sdim, int causal)
{
  __shared__ float red[64];
  __shared__ float srow[64];
  const int row = blockIdx.x;
  const int t   = row % Tq;
  const int b   = (row / Tq) % Bdim;
  const int s   = threadIdx.x;
  float v = -INFINITY;
#if SOFTMAX_ASYNC
  {
    const int si = (s < Sdim) ? s : (Sdim - 1);
    float* gp = S + (long long)row * Sdim + si;
    __builtin_amdgcn_global_load_async_to_lds_b32(
        (AS1 int*)gp, (AS3 int*)&srow[s], 0, 0);
    __builtin_amdgcn_s_wait_asynccnt(0);
    __syncthreads();
    if (s < Sdim) v = srow[s];
  }
#else
  if (s < Sdim) v = S[(long long)row * Sdim + s];
#endif
  if (s < Sdim) {
    if (causal && s > t) v = -INFINITY;
    if (mask && mask[b * Sdim + s]) v = -INFINITY;
  }
  red[s] = v; __syncthreads();
  for (int off = 32; off > 0; off >>= 1) {
    if (s < off) red[s] = fmaxf(red[s], red[s + off]);
    __syncthreads();
  }
  const float mx = red[0]; __syncthreads();
  float e = (s < Sdim && mx > -INFINITY) ? __expf(v - mx) : 0.0f;
  red[s] = e; __syncthreads();
  for (int off = 32; off > 0; off >>= 1) {
    if (s < off) red[s] += red[s + off];
    __syncthreads();
  }
  const float sum = red[0];
  if (s < Sdim) S[(long long)row * Sdim + s] = (sum > 0.0f) ? e / sum : 0.0f;
}

// out[row] = LN(Y[row] + R[row]) * g + beta   (biased variance, eps 1e-5)
__global__ void __launch_bounds__(256)
layernorm_res(const float* __restrict__ Y, const float* __restrict__ R,
              const float* __restrict__ g, const float* __restrict__ beta,
              float* __restrict__ out, int D)
{
  __shared__ float s1[256], s2[256];
  const int row = blockIdx.x;
  const int tid = threadIdx.x;
  float a = 0.0f, q = 0.0f;
  for (int d = tid; d < D; d += 256) {
    float v = Y[(long long)row * D + d] + (R ? R[(long long)row * D + d] : 0.0f);
    a += v; q += v * v;
  }
  s1[tid] = a; s2[tid] = q; __syncthreads();
  for (int off = 128; off > 0; off >>= 1) {
    if (tid < off) { s1[tid] += s1[tid + off]; s2[tid] += s2[tid + off]; }
    __syncthreads();
  }
  const float mu   = s1[0] / (float)D;
  const float var  = s2[0] / (float)D - mu * mu;
  const float rstd = rsqrtf(var + 1e-5f);
  for (int d = tid; d < D; d += 256) {
    float v = Y[(long long)row * D + d] + (R ? R[(long long)row * D + d] : 0.0f);
    out[(long long)row * D + d] = (v - mu) * rstd * g[d] + beta[d];
  }
}

// =====================================================================
// Host orchestration
// =====================================================================
struct MhaP { const float *Wq, *Wk, *Wv, *Wo, *bo; };
struct FfP  { const float *W1, *b1, *W2, *b2, *g, *b; };

extern "C" void kernel_launch(void* const* d_in, const int* in_sizes, int n_in,
                              void* d_out, int out_size, void* d_ws, size_t ws_size,
                              hipStream_t stream)
{
  (void)in_sizes; (void)n_in; (void)out_size; (void)ws_size;

  // ---- inputs in setup_inputs() flattened dict order ----
  const float*         conv_cls  = (const float*)d_in[0];
  const float*         persona   = (const float*)d_in[1];
  const float*         adj       = (const float*)d_in[2];
  const unsigned char* conv_mask = (const unsigned char*)d_in[3];
  const unsigned char* henc_mask = (const unsigned char*)d_in[4];
  const int*           dec_ids   = (const int*)d_in[5];
  const unsigned char* dec_mask  = (const unsigned char*)d_in[6];
  /* d_in[7] = labels (unused in forward) */
  const float*         pe        = (const float*)d_in[8];
  const float*         emb       = (const float*)d_in[9];
  const MhaP conv_mha { (const float*)d_in[10], (const float*)d_in[11], (const float*)d_in[12],
                        (const float*)d_in[13], (const float*)d_in[14] };
  const MhaP r_mha    { (const float*)d_in[15], (const float*)d_in[16], (const float*)d_in[17],
                        (const float*)d_in[18], (const float*)d_in[19] };
  const MhaP re_mha   { (const float*)d_in[20], (const float*)d_in[21], (const float*)d_in[22],
                        (const float*)d_in[23], (const float*)d_in[24] };
  const float* gcn1W = (const float*)d_in[25]; const float* gcn1b = (const float*)d_in[26];
  const float* gcn2W = (const float*)d_in[27]; const float* gcn2b = (const float*)d_in[28];
  const FfP gff { (const float*)d_in[29], (const float*)d_in[30], (const float*)d_in[31],
                  (const float*)d_in[32], (const float*)d_in[33], (const float*)d_in[34] };
  const FfP lff { (const float*)d_in[35], (const float*)d_in[36], (const float*)d_in[37],
                  (const float*)d_in[38], (const float*)d_in[39], (const float*)d_in[40] };
  const float* lm_head = (const float*)d_in[41];

  float* henc   = (float*)d_out;                               // [B*NN, D]
  float* logits = (float*)d_out + (size_t)B_ * NN_ * D_;       // [B*TD, V]

  // ---- workspace bump allocator (floats) ----
  float* ws = (float*)d_ws;
  size_t off = 0;
  auto alloc = [&](size_t n) { float* p = ws + off; off += n; return p; };
  float* X1  = alloc((size_t)B_ * TC_ * D_);       // conv input
  float* Qb  = alloc((size_t)H_ * 2048 * D_);      // per-head Q (max rows 2048)
  float* Kb  = alloc((size_t)H_ * 2048 * D_);
  float* Vb  = alloc((size_t)H_ * 2048 * D_);
  float* Sb  = alloc((size_t)H_ * B_ * 64 * 64);   // scores
  float* ATb = alloc((size_t)B_ * TC_ * H_ * D_);  // head-concat attention out
  float* X2  = alloc((size_t)2048 * D_);           // conv MHA projected out
  float* G   = alloc((size_t)2048 * D_);           // graph activations
  float* T1  = alloc((size_t)2048 * D_);           // adj @ x temp
  float* G2  = alloc((size_t)2048 * D_);           // gcn2 out (residual for graph_ff)
  float* F1  = alloc((size_t)2048 * 4 * D_);       // FF hidden
  float* F2  = alloc((size_t)2048 * D_);           // FF out (pre-LN)
  float* R   = alloc((size_t)B_ * TD_ * D_);       // decoder embeddings
  float* HR  = alloc((size_t)B_ * TD_ * D_);       // r_mha out
  float* O   = alloc((size_t)B_ * TD_ * D_);       // cross-attn out
  float* X3  = alloc((size_t)B_ * TD_ * D_);       // logits_ff out

  const int TPB = 256;
  auto gemm = [&](const float* A, const float* Bm, float* C, const float* bias,
                  int M, int N, int K, int lda, int ldb, int ldc,
                  int Z, int Z0, long long sA0, long long sA1,
                  long long sB0, long long sB1, long long sC0, long long sC1,
                  float alpha, int flags)
  {
    dim3 grid((unsigned)((N + BN - 1) / BN), (unsigned)((M + BM - 1) / BM), (unsigned)Z);
    gemm_bf16_wmma<<<grid, dim3(256), 0, stream>>>(A, Bm, C, bias, M, N, K, lda, ldb, ldc,
                                                   Z0, sA0, sA1, sB0, sB1, sC0, sC1,
                                                   alpha, flags);
  };

  // _mha: each head projects full D->D; heads batched on gridDim.z
  auto run_mha = [&](const float* qin, int Tq, const float* kvin, int Skv,
                     const MhaP& p, const unsigned char* mask, int causal, float* out)
  {
    const long long BTq = (long long)B_ * Tq, BSk = (long long)B_ * Skv;
    const float scale = 1.0f / sqrtf((float)D_);
    // Q/K/V projections: z = h; W stride D*D, out stride B*T*D
    gemm(qin,  p.Wq, Qb, nullptr, (int)BTq, D_, D_, D_, D_, D_, H_, H_,
         0, 0, (long long)D_ * D_, 0, BTq * D_, 0, 1.0f, 0);
    gemm(kvin, p.Wk, Kb, nullptr, (int)BSk, D_, D_, D_, D_, D_, H_, H_,
         0, 0, (long long)D_ * D_, 0, BSk * D_, 0, 1.0f, 0);
    gemm(kvin, p.Wv, Vb, nullptr, (int)BSk, D_, D_, D_, D_, D_, H_, H_,
         0, 0, (long long)D_ * D_, 0, BSk * D_, 0, 1.0f, 0);
    // scores[h][b] = Q[h,b] @ K[h,b]^T * scale   (z0=b, z1=h; B given as [S,K] -> BT)
    gemm(Qb, Kb, Sb, nullptr, Tq, Skv, D_, D_, D_, Skv, H_ * B_, B_,
         (long long)Tq * D_, BTq * D_, (long long)Skv * D_, BSk * D_,
         (long long)Tq * Skv, (long long)B_ * Tq * Skv, scale, 1);
    softmax_rows<<<H_ * B_ * Tq, 64, 0, stream>>>(Sb, mask, B_, Tq, Skv, causal);
    // out-head = wei @ V, written straight into concat layout [b][t][h*D + d]
    gemm(Sb, Vb, ATb, nullptr, Tq, D_, Skv, Skv, D_, H_ * D_, H_ * B_, B_,
         (long long)Tq * Skv, (long long)B_ * Tq * Skv, (long long)Skv * D_, BSk * D_,
         (long long)Tq * H_ * D_, (long long)D_, 1.0f, 0);
    // output projection [B*Tq, H*D] @ Wo + bo
    gemm(ATb, p.Wo, out, p.bo, (int)BTq, D_, H_ * D_, H_ * D_, D_, D_, 1, 1,
         0, 0, 0, 0, 0, 0, 1.0f, 0);
  };

  // ---- 1) conv branch: x = conv_cls + pe[TC] ----
  {
    const int tot = B_ * TC_ * D_;
    add_pe_row<<<(tot + TPB - 1) / TPB, TPB, 0, stream>>>(conv_cls, pe + (size_t)TC_ * D_, X1, tot, D_);
  }
  // ---- 2) conv self-attention (key mask, not causal) ----
  run_mha(X1, TC_, X1, TC_, conv_mha, conv_mask, 0, X2);
  // ---- 3) x = att * !mask; concat [persona, x] -> G [B, 64, D] ----
  {
    const int tot = B_ * NN_ * D_;
    concat_graph<<<(tot + TPB - 1) / TPB, TPB, 0, stream>>>(persona, X2, conv_mask, G, tot);
  }
  // ---- 4) GCN1: relu((adj@G)@W + b) ----
  gemm(adj, G, T1, nullptr, NN_, D_, NN_, NN_, D_, D_, B_, B_,
       (long long)NN_ * NN_, 0, (long long)NN_ * D_, 0, (long long)NN_ * D_, 0, 1.0f, 0);
  gemm(T1, gcn1W, G, gcn1b, B_ * NN_, D_, D_, D_, D_, D_, 1, 1, 0, 0, 0, 0, 0, 0, 1.0f, 2);
  // ---- 5) GCN2 ----
  gemm(adj, G, T1, nullptr, NN_, D_, NN_, NN_, D_, D_, B_, B_,
       (long long)NN_ * NN_, 0, (long long)NN_ * D_, 0, (long long)NN_ * D_, 0, 1.0f, 0);
  gemm(T1, gcn2W, G2, gcn2b, B_ * NN_, D_, D_, D_, D_, D_, 1, 1, 0, 0, 0, 0, 0, 0, 1.0f, 2);
  // ---- 6) graph_ff -> h_enc (into d_out) ----
  gemm(G2, gff.W1, F1, gff.b1, B_ * NN_, 4 * D_, D_, D_, 4 * D_, 4 * D_, 1, 1,
       0, 0, 0, 0, 0, 0, 1.0f, 2);
  gemm(F1, gff.W2, F2, gff.b2, B_ * NN_, D_, 4 * D_, 4 * D_, D_, D_, 1, 1,
       0, 0, 0, 0, 0, 0, 1.0f, 0);
  layernorm_res<<<B_ * NN_, 256, 0, stream>>>(F2, G2, gff.g, gff.b, henc, D_);
  // ---- 7) decoder embeddings: r = emb[ids] + pe[TD] ----
  {
    const int tot = B_ * TD_ * D_;
    embed_pe<<<(tot + TPB - 1) / TPB, TPB, 0, stream>>>(dec_ids, emb, pe + (size_t)TD_ * D_, R, tot, D_);
  }
  // ---- 8) decoder causal self-attention ----
  run_mha(R, TD_, R, TD_, r_mha, dec_mask, 1, HR);
  // ---- 9) cross-attention over h_enc ----
  run_mha(HR, TD_, henc, NN_, re_mha, henc_mask, 0, O);
  // ---- 10) logits_ff ----
  gemm(O, lff.W1, F1, lff.b1, B_ * TD_, 4 * D_, D_, D_, 4 * D_, 4 * D_, 1, 1,
       0, 0, 0, 0, 0, 0, 1.0f, 2);
  gemm(F1, lff.W2, F2, lff.b2, B_ * TD_, D_, 4 * D_, 4 * D_, D_, D_, 1, 1,
       0, 0, 0, 0, 0, 0, 1.0f, 0);
  layernorm_res<<<B_ * TD_, 256, 0, stream>>>(F2, O, lff.g, lff.b, X3, D_);
  // ---- 11) lm_head: [1024,768] @ [768,50257] (ragged N handled by GEMM bounds) ----
  gemm(X3, lm_head, logits, nullptr, B_ * TD_, V_, D_, D_, V_, V_, 1, 1,
       0, 0, 0, 0, 0, 0, 1.0f, 0);
}
